// graph_encoder_with_contrastive_32401233281584
// MI455X (gfx1250) — matrix operand order repaired and checked by reading, not compiled, not run on hardware
//
#include <hip/hip_runtime.h>
#include <hip/hip_bf16.h>
#include <math.h>

// ---------------- problem constants (match reference) ----------------
#define NN   8192
#define EE   131072
#define KDIM 3000
#define K1P  3008      // KDIM padded to a multiple of 32 for guard-free B fragments
#define H1   256
#define OF   64

typedef __attribute__((ext_vector_type(16))) __bf16 v16bf;
typedef __attribute__((ext_vector_type(8)))  __bf16 v8bf;
typedef __attribute__((ext_vector_type(8)))  float  v8f;

static __device__ __forceinline__ v8f wmma_bf16(v16bf a, v16bf b, v8f c) {
  // D = A(16x32 bf16) x B(32x16 bf16) + C(16x16 f32)
  return __builtin_amdgcn_wmma_f32_16x16x32_bf16(false, a, false, b, (short)0, c,
                                                 false, false);
}

static __device__ __forceinline__ unsigned pk_bf16(float lo, float hi) {
  union { __bf16 b[2]; unsigned u; } t;
  t.b[0] = (__bf16)lo; t.b[1] = (__bf16)hi;   // -> v_cvt_pk_bf16_f32
  return t.u;
}

static __device__ __forceinline__ v16bf cat8(v8bf lo, v8bf hi) {
  return __builtin_shufflevector(lo, hi, 0, 1, 2, 3, 4, 5, 6, 7,
                                 8, 9, 10, 11, 12, 13, 14, 15);
}

// 16x16 f32 C/D tile store: VGPR r -> (M=r, N=lane) lanes 0-15; (M=r+8, N=lane-16) lanes 16-31
static __device__ __forceinline__ void store_c(float* C, int ldc, int m0, int n0, v8f c) {
  int lane = threadIdx.x & 31;
  int n = lane & 15;
  int mo = (lane >> 4) << 3;
#pragma unroll
  for (int r = 0; r < 8; ++r)
    C[(size_t)(m0 + mo + r) * ldc + (n0 + n)] = c[r];
}

// ---------------- W1 -> transposed, zero-padded bf16 W1T[256][3008] ----------------
__global__ void k_cvtW1(const float* __restrict__ W1, __bf16* __restrict__ W1T) {
  int idx = blockIdx.x * blockDim.x + threadIdx.x;   // H1 * K1P
  if (idx >= H1 * K1P) return;
  int c = idx / K1P, k = idx - c * K1P;
  W1T[idx] = (k < KDIM) ? (__bf16)W1[(size_t)k * H1 + c] : (__bf16)0.f;
}

// ---------------- GEMM1: xw1 = gene[8192x3000] @ W1[3000x256] ----------------
// A staged via LDS (packed bf16 pairs, fragments = 2x ds_load_b128);
// B fragments come straight from W1T as 2x global 16B loads (L2-resident, padded).
__global__ void __launch_bounds__(256) k_gemm1(const float* __restrict__ A,
                                               const __bf16* __restrict__ BT,
                                               float* __restrict__ C) {
  __shared__ __bf16 sA[16 * 32];   // [r][k]
  const int tid  = threadIdx.x;
  const int lane = tid & 31;
  const int wave = tid >> 5;              // 8 waves x 32 output cols
  const int mt   = blockIdx.x * 16;
  const int r    = lane & 15;
  const int klo  = (lane >> 4) << 4;      // A layout: 0/16
  const int kb   = (lane >> 4) << 4;      // B layout: 0/16
  const int col0 = wave * 32 + r;
  unsigned* sA32 = (unsigned*)sA;

  const int arow = tid >> 4;              // staging row
  const int akk  = (tid & 15) << 1;       // staging k-pair
  const float*  aptr = A + (size_t)(mt + arow) * KDIM + akk;
  const __bf16* b0p  = BT + (size_t)col0 * K1P + kb;
  const __bf16* b1p  = BT + (size_t)(col0 + 16) * K1P + kb;

  v8f c0 = {}; v8f c1 = {};
  int kc = 0;
  // ---- full tiles: no guards anywhere (93 iterations)
  for (; kc + 32 <= KDIM; kc += 32) {
    const float2 v = *(const float2*)(aptr + kc);
    sA32[(arow << 4) | (akk >> 1)] = pk_bf16(v.x, v.y);
    if (kc + 32 + 32 <= KDIM) __builtin_prefetch(aptr + kc + 32, 0, 3);
    __syncthreads();
    v16bf a  = cat8(*(const v8bf*)&sA[(r << 5) + klo],
                    *(const v8bf*)&sA[(r << 5) + klo + 8]);
    v16bf b0 = cat8(*(const v8bf*)(b0p + kc), *(const v8bf*)(b0p + kc + 8));
    v16bf b1 = cat8(*(const v8bf*)(b1p + kc), *(const v8bf*)(b1p + kc + 8));
    c0 = wmma_bf16(a, b0, c0);
    c1 = wmma_bf16(a, b1, c1);
    __syncthreads();
  }
  // ---- guarded tail (kc == 2976, 24 valid k; B side reads zero padding)
  if (kc < KDIM) {
    float2 v = make_float2(0.f, 0.f);
    if (kc + akk < KDIM) v = *(const float2*)(aptr + kc);   // KDIM even, pair never straddles
    sA32[(arow << 4) | (akk >> 1)] = pk_bf16(v.x, v.y);
    __syncthreads();
    v16bf a  = cat8(*(const v8bf*)&sA[(r << 5) + klo],
                    *(const v8bf*)&sA[(r << 5) + klo + 8]);
    v16bf b0 = cat8(*(const v8bf*)(b0p + kc), *(const v8bf*)(b0p + kc + 8));
    v16bf b1 = cat8(*(const v8bf*)(b1p + kc), *(const v8bf*)(b1p + kc + 8));
    c0 = wmma_bf16(a, b0, c0);
    c1 = wmma_bf16(a, b1, c1);
  }
  store_c(C, H1, mt, wave * 32, c0);
  store_c(C, H1, mt, wave * 32 + 16, c1);
}

// ---------------- transpose W2 -> bf16 W2T[64][256] (fragment-contiguous B) ----------------
__global__ void k_cvtW2(const float* __restrict__ W2, __bf16* __restrict__ W2T) {
  int idx = blockIdx.x * blockDim.x + threadIdx.x;   // 64*256
  int c = idx >> 8, k = idx & 255;
  W2T[idx] = (__bf16)W2[(size_t)k * OF + c];
}

// ---------------- GEMM2: z = hidden[8192x256] @ W2[256x64], both graphs via grid.y ----------
__global__ void __launch_bounds__(128) k_gemm2(const float* __restrict__ Hm,
                                               const float* __restrict__ Hmc,
                                               const __bf16* __restrict__ W2T,
                                               float* __restrict__ Z,
                                               float* __restrict__ Zc) {
  __shared__ __bf16 sA[16 * 32];
  const float* A = blockIdx.y ? Hmc : Hm;
  float*       C = blockIdx.y ? Zc  : Z;
  const int tid = threadIdx.x, lane = tid & 31, wave = tid >> 5;  // 4 waves x 16 cols
  const int mt  = blockIdx.x * 16;
  const int r   = lane & 15;
  const int klo = (lane >> 4) << 4;
  const int kb  = (lane >> 4) << 4;
  const int col = wave * 16 + r;
  unsigned* sA32 = (unsigned*)sA;
  v8f c = {};
  for (int kc = 0; kc < H1; kc += 32) {
    // stage A: 256 pairs / 128 threads
#pragma unroll
    for (int it = 0; it < 2; ++it) {
      int e  = tid + it * 128;
      int rr = e >> 4, kk = (e & 15) << 1;
      const float2 v = *(const float2*)(A + (size_t)(mt + rr) * H1 + kc + kk);
      sA32[(rr << 4) | (kk >> 1)] = pk_bf16(v.x, v.y);
    }
    __syncthreads();
    v16bf a = cat8(*(const v8bf*)&sA[(r << 5) + klo],
                   *(const v8bf*)&sA[(r << 5) + klo + 8]);
    const __bf16* bp = W2T + (size_t)col * H1 + kc + kb;
    v16bf b = cat8(*(const v8bf*)bp, *(const v8bf*)(bp + 8));   // global b128 x2
    c = wmma_bf16(a, b, c);
    __syncthreads();
  }
  store_c(C, OF, mt, wave * 16, c);
}

// ---------------- readout: [vsum|vsumc] = mask @ [x1|x1c] + fused row-sum ----------------
// B pre-transposed in bf16: XT[128][8192] (cols 0..63 = x1^T, 64..127 = x1c^T).
// Mask staged via LDS; row-sum accumulated exactly (f32) by the staging threads.
__global__ void __launch_bounds__(256) k_readout(const float* __restrict__ Mmask,
                                                 const __bf16* __restrict__ XT,
                                                 float* __restrict__ Vs,
                                                 float* __restrict__ Vsc,
                                                 float* __restrict__ Rs) {
  __shared__ __bf16 sA[16 * 32];
  const int tid = threadIdx.x, lane = tid & 31, wave = tid >> 5;  // 8 waves x 16 cols
  const int mt  = blockIdx.x * 16;
  const int r   = lane & 15;
  const int klo = (lane >> 4) << 4;
  const int kb  = (lane >> 4) << 4;
  const int gcol = wave * 16 + r;          // global column into XT (0..127)
  float* Csel   = (wave < 4) ? Vs : Vsc;
  const int nb  = (wave < 4) ? wave * 16 : (wave - 4) * 16;
  unsigned* sA32 = (unsigned*)sA;

  const int arow = tid >> 4;
  const int akk  = (tid & 15) << 1;
  const float* aptr = Mmask + (size_t)(mt + arow) * NN + akk;
  const __bf16* bp  = XT + (size_t)gcol * NN + kb;

  float rsum = 0.f;                        // partial row-sum of row (mt+arow)
  v8f c = {};
  for (int kc = 0; kc < NN; kc += 32) {
    {
      const float2 v = *(const float2*)(aptr + kc);
      rsum += v.x + v.y;                   // exact: mask entries are 0/1
      sA32[(arow << 4) | (akk >> 1)] = pk_bf16(v.x, v.y);
    }
    __syncthreads();
    v16bf a = cat8(*(const v8bf*)&sA[(r << 5) + klo],
                   *(const v8bf*)&sA[(r << 5) + klo + 8]);
    v16bf b = cat8(*(const v8bf*)(bp + kc), *(const v8bf*)(bp + kc + 8));
    c = wmma_bf16(a, b, c);
    __syncthreads();
  }
  store_c(Csel, OF, mt, nb, c);
  // reduce 16 staging partials of each row (threads tid = arow*16 + p)
  rsum += __shfl_xor(rsum, 1);
  rsum += __shfl_xor(rsum, 2);
  rsum += __shfl_xor(rsum, 4);
  rsum += __shfl_xor(rsum, 8);
  if ((tid & 15) == 0) Rs[mt + arow] = rsum;
}

// ---------------- small elementwise / graph kernels ----------------
__global__ void k_fill(float* p, float v, int n) {
  int i = blockIdx.x * blockDim.x + threadIdx.x;
  if (i < n) p[i] = v;
}
__global__ void k_degcount(float* deg, const int* __restrict__ dst, int e) {
  int i = blockIdx.x * blockDim.x + threadIdx.x;
  if (i < e) atomicAdd(&deg[dst[i]], 1.f);
}
__global__ void k_dinv(float* dinv, const float* __restrict__ deg, int n) {
  int i = blockIdx.x * blockDim.x + threadIdx.x;
  if (i < n) dinv[i] = rsqrtf(deg[i]);
}
// layer-1 edge scatter; corrupted graph reuses xw1 via perm (X[perm]@W1 == (X@W1)[perm])
__global__ void __launch_bounds__(256) k_agg1(const float* __restrict__ xw,
                                              const int* __restrict__ src,
                                              const int* __restrict__ dst,
                                              const int* __restrict__ perm,
                                              const float* __restrict__ dinv,
                                              float* agg, float* aggc) {
  int e = blockIdx.x;
  int j = src[e], i = dst[e];
  float w = dinv[i] * dinv[j];
  int f = threadIdx.x;
  atomicAdd(&agg[(size_t)i * H1 + f],  w * xw[(size_t)j * H1 + f]);
  atomicAdd(&aggc[(size_t)i * H1 + f], w * xw[(size_t)perm[j] * H1 + f]);
}
__global__ void k_fin1(float* agg, float* aggc, const float* __restrict__ xw,
                       const int* __restrict__ perm, const float* __restrict__ dinv,
                       const float* __restrict__ b1) {
  int idx = blockIdx.x * blockDim.x + threadIdx.x;
  int i = idx >> 8, f = idx & 255;
  float w = dinv[i]; w *= w;                       // self-loop norm
  float v  = agg[idx]  + w * xw[(size_t)i * H1 + f]       + b1[f];
  float vc = aggc[idx] + w * xw[(size_t)perm[i] * H1 + f] + b1[f];
  agg[idx]  = fmaxf(v, 0.f);
  aggc[idx] = fmaxf(vc, 0.f);
}
__global__ void __launch_bounds__(256) k_agg2(const float* __restrict__ z,
                                              const float* __restrict__ zc,
                                              const int* __restrict__ src,
                                              const int* __restrict__ dst,
                                              const float* __restrict__ dinv,
                                              float* agg, float* aggc) {
  int t = blockIdx.x * blockDim.x + threadIdx.x;   // 4 edges per block
  int e = t >> 6, f = t & 63;
  int j = src[e], i = dst[e];
  float w = dinv[i] * dinv[j];
  atomicAdd(&agg[(size_t)i * OF + f],  w * z[(size_t)j * OF + f]);
  atomicAdd(&aggc[(size_t)i * OF + f], w * zc[(size_t)j * OF + f]);
}
// layer-2 finalize: relu + bias + self-loop; also emit the bf16 transposed copies for readout
__global__ void k_fin2(const float* __restrict__ agg, const float* __restrict__ aggc,
                       const float* __restrict__ z, const float* __restrict__ zc,
                       const float* __restrict__ dinv, const float* __restrict__ b2,
                       float* x1, float* x1c, __bf16* __restrict__ XT) {
  int idx = blockIdx.x * blockDim.x + threadIdx.x;
  int i = idx >> 6, f = idx & 63;
  float w = dinv[i]; w *= w;
  float a = fmaxf(agg[idx]  + w * z[idx]  + b2[f], 0.f);
  float b = fmaxf(aggc[idx] + w * zc[idx] + b2[f], 0.f);
  x1[idx]  = a;
  x1c[idx] = b;
  XT[(size_t)f * NN + i]        = (__bf16)a;   // x1^T   (cols 0..63)
  XT[(size_t)(OF + f) * NN + i] = (__bf16)b;   // x1c^T  (cols 64..127)
}

// ---------------- normalize + sigmoid + bilinear head (one block per node) ----------------
__global__ void __launch_bounds__(64) k_bilinear(const float* __restrict__ X1,
                                                 const float* __restrict__ X1c,
                                                 const float* __restrict__ Vs,
                                                 const float* __restrict__ Vsc,
                                                 const float* __restrict__ Rs,
                                                 const float* __restrict__ Wd,
                                                 const float* __restrict__ bd,
                                                 float* ret1, float* ret1c) {
  __shared__ float sh[64];
  __shared__ float g[64], gc[64];
  const int n = blockIdx.x, f = threadIdx.x;
  const float rs = Rs[n];
  float t  = Vs[(size_t)n * OF + f]  / rs;
  float tc = Vsc[(size_t)n * OF + f] / rs;

  sh[f] = t * t; __syncthreads();
  for (int s = 32; s > 0; s >>= 1) { if (f < s) sh[f] += sh[f + s]; __syncthreads(); }
  float nrm = fmaxf(sqrtf(sh[0]), 1e-12f); __syncthreads();
  g[f] = 1.f / (1.f + expf(-(t / nrm)));

  sh[f] = tc * tc; __syncthreads();
  for (int s = 32; s > 0; s >>= 1) { if (f < s) sh[f] += sh[f + s]; __syncthreads(); }
  float nrmc = fmaxf(sqrtf(sh[0]), 1e-12f); __syncthreads();
  gc[f] = 1.f / (1.f + expf(-(tc / nrmc)));
  __syncthreads();

  // t1[i] = sum_j Wd[0,i,j] * g[j]
  float t1 = 0.f, t1c = 0.f;
  for (int j = 0; j < 64; ++j) {
    float w = Wd[f * 64 + j];
    t1 += w * g[j]; t1c += w * gc[j];
  }
  const float h  = X1[(size_t)n * OF + f];
  const float hc = X1c[(size_t)n * OF + f];
  float d11, d21, d22, d12;
  sh[f] = h * t1; __syncthreads();
  for (int s = 32; s > 0; s >>= 1) { if (f < s) sh[f] += sh[f + s]; __syncthreads(); }
  d11 = sh[0]; __syncthreads();
  sh[f] = hc * t1; __syncthreads();
  for (int s = 32; s > 0; s >>= 1) { if (f < s) sh[f] += sh[f + s]; __syncthreads(); }
  d21 = sh[0]; __syncthreads();
  sh[f] = hc * t1c; __syncthreads();
  for (int s = 32; s > 0; s >>= 1) { if (f < s) sh[f] += sh[f + s]; __syncthreads(); }
  d22 = sh[0]; __syncthreads();
  sh[f] = h * t1c; __syncthreads();
  for (int s = 32; s > 0; s >>= 1) { if (f < s) sh[f] += sh[f + s]; __syncthreads(); }
  d12 = sh[0];
  if (f == 0) {
    float b = bd[0];
    ret1[n * 2 + 0]  = d11 + b;   // bilinear(x1,  g1)
    ret1[n * 2 + 1]  = d21 + b;   // bilinear(x1c, g1)
    ret1c[n * 2 + 0] = d22 + b;   // bilinear(x1c, g1c)
    ret1c[n * 2 + 1] = d12 + b;   // bilinear(x1,  g1c)
  }
}

// ---------------- host orchestration ----------------
extern "C" void kernel_launch(void* const* d_in, const int* in_sizes, int n_in,
                              void* d_out, int out_size, void* d_ws, size_t ws_size,
                              hipStream_t stream) {
  const float* gene = (const float*)d_in[0];
  const float* mask = (const float*)d_in[1];
  const float* W1   = (const float*)d_in[2];
  const float* b1   = (const float*)d_in[3];
  const float* W2   = (const float*)d_in[4];
  const float* b2   = (const float*)d_in[5];
  const float* Wd   = (const float*)d_in[6];
  const float* bd   = (const float*)d_in[7];
  const int*   ei   = (const int*)d_in[8];   // [2, E]
  const int*   perm = (const int*)d_in[9];
  const int* src = ei;
  const int* dst = ei + EE;

  float* out_x1 = (float*)d_out;             // [N, 64]
  float* ret1   = out_x1 + (size_t)NN * OF;  // [N, 2]
  float* ret1c  = ret1 + (size_t)NN * 2;     // [N, 2]

  float* w      = (float*)d_ws;
  float* deg    = w; w += NN;
  float* dinv   = w; w += NN;
  float* xw1    = w; w += (size_t)NN * H1;
  float* agg1   = w; w += (size_t)NN * H1;   // agg1/agg1c contiguous for one zero pass
  float* agg1c  = w; w += (size_t)NN * H1;
  float* z2     = w; w += (size_t)NN * OF;
  float* z2c    = w; w += (size_t)NN * OF;
  float* agg2   = w; w += (size_t)NN * OF;   // agg2/agg2c contiguous
  float* agg2c  = w; w += (size_t)NN * OF;
  float* x1c    = w; w += (size_t)NN * OF;
  float* vsum   = w; w += (size_t)NN * OF;
  float* vsumc  = w; w += (size_t)NN * OF;
  float* rowsum = w; w += NN;
  __bf16* W2T = (__bf16*)w;                  // [64][256]   (16B-aligned)
  __bf16* XT  = W2T + (size_t)OF * H1;       // [128][8192] (16B-aligned)
  __bf16* W1T = XT + (size_t)2 * OF * NN;    // [256][3008] (16B-aligned)
  (void)n_in; (void)in_sizes; (void)out_size; (void)ws_size;

  // re-initialize accumulators every call (graph-replay safe, no static state)
  k_fill<<<NN / 256, 256, 0, stream>>>(deg, 1.f, NN);                        // self-loop degree
  k_fill<<<(NN * H1 * 2) / 256, 256, 0, stream>>>(agg1, 0.f, NN * H1 * 2);
  k_fill<<<(NN * OF * 2) / 256, 256, 0, stream>>>(agg2, 0.f, NN * OF * 2);

  k_degcount<<<EE / 256, 256, 0, stream>>>(deg, dst, EE);
  k_dinv<<<NN / 256, 256, 0, stream>>>(dinv, deg, NN);
  k_cvtW1<<<(H1 * K1P) / 256, 256, 0, stream>>>(W1, W1T);
  k_cvtW2<<<(OF * H1) / 256, 256, 0, stream>>>(W2, W2T);

  // layer 1: shared GEMM (perm commutes with X@W1), then edge scatter for both graphs
  k_gemm1<<<NN / 16, 256, 0, stream>>>(gene, W1T, xw1);
  k_agg1<<<EE, 256, 0, stream>>>(xw1, src, dst, perm, dinv, agg1, agg1c);
  k_fin1<<<(NN * H1) / 256, 256, 0, stream>>>(agg1, agg1c, xw1, perm, dinv, b1);

  // layer 2
  k_gemm2<<<dim3(NN / 16, 2), 128, 0, stream>>>(agg1, agg1c, W2T, z2, z2c);
  k_agg2<<<(EE * OF) / 256, 256, 0, stream>>>(z2, z2c, src, dst, dinv, agg2, agg2c);
  k_fin2<<<(NN * OF) / 256, 256, 0, stream>>>(agg2, agg2c, z2, z2c, dinv, b2,
                                              out_x1, x1c, XT);

  // fused readout GEMM (single mask pass, fused row-sum) + bilinear head
  k_readout<<<NN / 16, 256, 0, stream>>>(mask, XT, vsum, vsumc, rowsum);
  k_bilinear<<<NN, 64, 0, stream>>>(out_x1, x1c, vsum, vsumc, rowsum, Wd, bd, ret1, ret1c);
}
